// Head_45183055954532
// MI455X (gfx1250) — compile-verified, matrix operand order checked by reading, not run on hardware
//
#include <hip/hip_runtime.h>
#include <hip/hip_bf16.h>

#define BB 8
#define TT 2048
#define CC 1024
#define HS 64

typedef __attribute__((ext_vector_type(16))) __bf16 v16bf;
typedef __attribute__((ext_vector_type(8)))  __bf16 v8bf;
typedef __attribute__((ext_vector_type(8)))  float  v8f;

__device__ __forceinline__ v16bf cat8(v8bf lo, v8bf hi) {
    return __builtin_shufflevector(lo, hi, 0,1,2,3,4,5,6,7,8,9,10,11,12,13,14,15);
}

// gfx1250 async DMA: global -> LDS, 16 bytes per lane, tracked by ASYNCcnt.
__device__ __forceinline__ void async_copy_b128(unsigned lds_byte_addr, const void* gptr) {
    asm volatile("global_load_async_to_lds_b128 %0, %1, off"
                 :: "v"(lds_byte_addr), "v"(gptr) : "memory");
}
__device__ __forceinline__ void wait_asynccnt0() {
    asm volatile("s_wait_asynccnt 0x0" ::: "memory");
}
__device__ __forceinline__ unsigned lds_addr32(const void* p) {
    // generic pointers to LDS carry the workgroup-relative LDS offset in [31:0]
    return (unsigned)(unsigned long long)(uintptr_t)p;
}

// ---------------------------------------------------------------------------
// Kernel 0: one-shot f32 -> bf16 conversion of the three weight matrices
// (0.75 MB -> 384 KB, stays L2-resident for the whole projection pass).
// ---------------------------------------------------------------------------
__global__ __launch_bounds__(256)
void wconv_kernel(const float* __restrict__ Wk, const float* __restrict__ Wq,
                  const float* __restrict__ Wv, __bf16* __restrict__ wbf)
{
    const int N   = HS * CC;                          // 65536 per matrix
    int idx = blockIdx.x * 256 + threadIdx.x;         // group of 8 elements
    int mat = idx / (N / 8);
    int off = (idx % (N / 8)) * 8;
    const float* src = (mat == 0) ? Wk : (mat == 1) ? Wq : Wv;
    __bf16*      dst = wbf + (size_t)mat * N;
#pragma unroll
    for (int e = 0; e < 8; ++e) dst[off + e] = (__bf16)src[off + e];
}

// ---------------------------------------------------------------------------
// Kernel 1: fused Q/K/V projection.  x[16384,1024] f32 -> q,k,v [16384,64] bf16
// One block = 16 rows of x staged once in LDS as bf16; 12 waves = 3 matrices x
// 4 column tiles; 32 x v_wmma_f32_16x16x32_bf16 over K=1024 per wave; both
// WMMA operands are 16-byte bf16 loads (no converts in the hot loop).
// ---------------------------------------------------------------------------
__global__ __launch_bounds__(384)
void proj_kernel(const float* __restrict__ x,
                 const __bf16* __restrict__ wbf,
                 __bf16* __restrict__ kout,
                 __bf16* __restrict__ qout,
                 __bf16* __restrict__ vout)
{
    __shared__ __align__(16) __bf16 xs[16 * CC];   // 32 KB

    const int  tid  = threadIdx.x;
    const long row0 = (long)blockIdx.x * 16;

    for (int idx = tid; idx < 16 * CC; idx += 384)
        xs[idx] = (__bf16)x[row0 * CC + idx];
    __syncthreads();

    const int wave = tid >> 5;
    const int lane = tid & 31;
    const int half = lane >> 4;
    const int l15  = lane & 15;

    const int mat = wave >> 2;            // 0=K, 1=Q, 2=V  (input order Wk,Wq,Wv)
    const int n0  = (wave & 3) * 16;

    __bf16* outp = (mat == 0) ? kout : (mat == 1 ? qout : vout);

    const __bf16* wrow = wbf + (size_t)mat * HS * CC + (size_t)(n0 + l15) * CC;
    const __bf16* xrow = xs + l15 * CC;

    v8f acc = {};
    for (int kb = 0; kb < CC; kb += 32) {
        v16bf A  = cat8(*(const v8bf*)(xrow + kb + half * 8),
                        *(const v8bf*)(xrow + kb + 16 + half * 8));
        v16bf Bm = cat8(*(const v8bf*)(wrow + kb + half * 8),
                        *(const v8bf*)(wrow + kb + 16 + half * 8));
        acc = __builtin_amdgcn_wmma_f32_16x16x32_bf16(false, A, false, Bm,
                                                      (short)0, acc, false, false);
    }

#pragma unroll
    for (int r = 0; r < 8; ++r) {
        long m = row0 + r + 8 * half;
        outp[m * HS + n0 + l15] = (__bf16)acc[r];
    }
}

// ---------------------------------------------------------------------------
// Kernel 2: flash attention with online softmax.  Block = 8 waves = 128 query
// rows of one batch.  Per 32-key step: K and V tiles DMA'd straight into LDS
// with GLOBAL_LOAD_ASYNC_TO_LDS_B128 (ASYNCcnt), V transposed in LDS, then
// 8 WMMA per wave (4 for S=QK^T, 4 for O+=P*V).
// ---------------------------------------------------------------------------
__global__ __launch_bounds__(256)
void attn_kernel(const __bf16* __restrict__ qg,
                 const __bf16* __restrict__ kg,
                 const __bf16* __restrict__ vg,
                 float* __restrict__ out)
{
    __shared__ __align__(16) __bf16 lds_k[32 * HS];      // [key][h]   4 KB
    __shared__ __align__(16) __bf16 lds_v[32 * HS];      // [key][h]   4 KB
    __shared__ __align__(16) __bf16 lds_vt[HS * 32];     // [h][key]   4 KB
    __shared__ __align__(16) __bf16 lds_p[8][16 * 32];   // per-wave P 8 KB

    const int tid  = threadIdx.x;
    const int wave = tid >> 5;
    const int lane = tid & 31;
    const int half = lane >> 4;
    const int l15  = lane & 15;

    const int b     = blockIdx.y;
    const int r0    = blockIdx.x * 128;
    const int qrow0 = r0 + wave * 16;

    const __bf16* qbase = qg + (long)b * TT * HS;
    const __bf16* kbase = kg + (long)b * TT * HS;
    const __bf16* vbase = vg + (long)b * TT * HS;

    // Q tile in A-operand layout, pre-scaled by C^-0.5 = 1/32
    const float scale = 0.03125f;
    v16bf aq0, aq1;
    {
        const __bf16* qrow = qbase + (long)(qrow0 + l15) * HS;
#pragma unroll
        for (int e = 0; e < 8; ++e) {
            aq0[e]     = (__bf16)((float)qrow[half * 8 + e]      * scale);
            aq0[e + 8] = (__bf16)((float)qrow[16 + half * 8 + e] * scale);
            aq1[e]     = (__bf16)((float)qrow[32 + half * 8 + e] * scale);
            aq1[e + 8] = (__bf16)((float)qrow[48 + half * 8 + e] * scale);
        }
    }

    v8f o[4] = {v8f{}, v8f{}, v8f{}, v8f{}};
    float mrow[8], lrow[8];
#pragma unroll
    for (int r = 0; r < 8; ++r) { mrow[r] = -1e30f; lrow[r] = 0.0f; }

    const unsigned ldsk_dst = lds_addr32(lds_k) + tid * 16u;
    const unsigned ldsv_dst = lds_addr32(lds_v) + tid * 16u;

    const int jend = r0 + 128;                 // causal: keys needed by last row
    for (int j0 = 0; j0 < jend; j0 += 32) {
        __syncthreads();                       // previous step's compute done
        // ---- async DMA: K and V 32x64 bf16 tiles straight into LDS ----
        const __bf16* gk = kbase + (long)j0 * HS + tid * 8;
        const __bf16* gv = vbase + (long)j0 * HS + tid * 8;
        async_copy_b128(ldsk_dst, gk);
        async_copy_b128(ldsv_dst, gv);
        if (j0 + 32 < jend) {                  // prefetch next tiles (in-bounds)
            __builtin_prefetch(gk + 32 * HS, 0, 0);
            __builtin_prefetch(gv + 32 * HS, 0, 0);
        }
        wait_asynccnt0();                      // this wave's DMA landed in LDS
        __syncthreads();                       // all waves' tiles visible
        // ---- transpose V in LDS: [key][h] -> [h][key] ----
        {
            int key = tid >> 3;                // 0..31
            int h0  = (tid & 7) * 8;           // 0..56
#pragma unroll
            for (int e = 0; e < 8; ++e)
                lds_vt[(h0 + e) * 32 + key] = lds_v[key * HS + h0 + e];
        }
        __syncthreads();

        if (j0 <= qrow0 + 15) {                // wave-uniform guard (EXEC full)
            // ---- S = Q K^T for two 16-key subtiles ----
            const __bf16* kr0 = lds_k + l15 * HS;
            const __bf16* kr1 = lds_k + (16 + l15) * HS;
            v16bf b00 = cat8(*(const v8bf*)(kr0 + half * 8),      *(const v8bf*)(kr0 + 16 + half * 8));
            v16bf b01 = cat8(*(const v8bf*)(kr0 + 32 + half * 8), *(const v8bf*)(kr0 + 48 + half * 8));
            v16bf b10 = cat8(*(const v8bf*)(kr1 + half * 8),      *(const v8bf*)(kr1 + 16 + half * 8));
            v16bf b11 = cat8(*(const v8bf*)(kr1 + 32 + half * 8), *(const v8bf*)(kr1 + 48 + half * 8));
            v8f s0 = {}, s1 = {};
            s0 = __builtin_amdgcn_wmma_f32_16x16x32_bf16(false, aq0, false, b00, (short)0, s0, false, false);
            s0 = __builtin_amdgcn_wmma_f32_16x16x32_bf16(false, aq1, false, b01, (short)0, s0, false, false);
            s1 = __builtin_amdgcn_wmma_f32_16x16x32_bf16(false, aq0, false, b10, (short)0, s1, false, false);
            s1 = __builtin_amdgcn_wmma_f32_16x16x32_bf16(false, aq1, false, b11, (short)0, s1, false, false);

            // ---- causal mask + online softmax (rows align with C layout) ----
#pragma unroll
            for (int r = 0; r < 8; ++r) {
                int rowg = qrow0 + r + 8 * half;
                float v0 = ((j0 + l15)      > rowg) ? -1e30f : s0[r];
                float v1 = ((j0 + 16 + l15) > rowg) ? -1e30f : s1[r];
                float mx = fmaxf(v0, v1);
#pragma unroll
                for (int off = 1; off < 16; off <<= 1)
                    mx = fmaxf(mx, __shfl_xor(mx, off, 32));
                float mnew  = fmaxf(mrow[r], mx);
                float alpha = __expf(mrow[r] - mnew);
                float p0 = __expf(v0 - mnew);
                float p1 = __expf(v1 - mnew);
                float ps = p0 + p1;
#pragma unroll
                for (int off = 1; off < 16; off <<= 1)
                    ps += __shfl_xor(ps, off, 32);
                lrow[r] = lrow[r] * alpha + ps;
                mrow[r] = mnew;
#pragma unroll
                for (int t = 0; t < 4; ++t) o[t][r] *= alpha;
                // park P in per-wave LDS ([row][key]) for A-layout repack
                __bf16* prow = &lds_p[wave][(r + 8 * half) * 32];
                prow[l15]      = (__bf16)p0;
                prow[16 + l15] = (__bf16)p1;
            }
            // in-wave cross-lane LDS RAW: DS ops are in-order, drain DScnt
            asm volatile("s_wait_dscnt 0" ::: "memory");

            // ---- O += P @ V : A = P (16x32), B = V^T columns ----
            const __bf16* prow = &lds_p[wave][l15 * 32];
            v16bf ap = cat8(*(const v8bf*)(prow + half * 8),
                            *(const v8bf*)(prow + 16 + half * 8));
#pragma unroll
            for (int t = 0; t < 4; ++t) {
                const __bf16* vcol = lds_vt + (t * 16 + l15) * 32;
                v16bf bv = cat8(*(const v8bf*)(vcol + half * 8),
                                *(const v8bf*)(vcol + 16 + half * 8));
                o[t] = __builtin_amdgcn_wmma_f32_16x16x32_bf16(false, ap, false, bv,
                                                               (short)0, o[t], false, false);
            }
        }
    }

    // ---- epilogue: normalize by row sums, store f32 ----
    float* obase = out + ((long)b * TT + qrow0) * HS;
#pragma unroll
    for (int r = 0; r < 8; ++r) {
        float inv = 1.0f / lrow[r];
        long  m   = (long)(r + 8 * half) * HS;
#pragma unroll
        for (int t = 0; t < 4; ++t)
            obase[m + t * 16 + l15] = o[t][r] * inv;
    }
}

// ---------------------------------------------------------------------------
extern "C" void kernel_launch(void* const* d_in, const int* in_sizes, int n_in,
                              void* d_out, int out_size, void* d_ws, size_t ws_size,
                              hipStream_t stream)
{
    const float* x  = (const float*)d_in[0];
    const float* Wk = (const float*)d_in[1];
    const float* Wq = (const float*)d_in[2];
    const float* Wv = (const float*)d_in[3];
    float*       out = (float*)d_out;

    const size_t NQKV = (size_t)BB * TT * HS;          // 1M elements, 2 MB each
    __bf16* kws = (__bf16*)d_ws;
    __bf16* qws = kws + NQKV;
    __bf16* vws = qws + NQKV;
    __bf16* wbf = vws + NQKV;                          // 3 * 64 * 1024 bf16

    wconv_kernel<<<dim3(3 * HS * CC / 8 / 256), 256, 0, stream>>>(Wk, Wq, Wv, wbf);
    proj_kernel<<<dim3(BB * TT / 16), 384, 0, stream>>>(x, wbf, kws, qws, vws);
    attn_kernel<<<dim3(TT / 128, BB), 256, 0, stream>>>(qws, kws, vws, out);
}
